// EmbeddingAndSeqformer_50946902065204
// MI455X (gfx1250) — compile-verified, matrix operand order checked by reading, not run on hardware
//
#include <hip/hip_runtime.h>

// ---------------------------------------------------------------------------
// Types for CDNA5 WMMA (wave32)
// ---------------------------------------------------------------------------
typedef __attribute__((ext_vector_type(16))) _Float16 v16h;
typedef __attribute__((ext_vector_type(8)))  float    v8f;

#define LSEQ   256
#define SEQC   384
#define PAIRC  128
#define HSEQ   8
#define HPAIR  4
#define DSEQ   48      // SEQC / HSEQ
#define DPAIR  32      // PAIRC / HPAIR
#define OPMH   32
#define MAXREL 32

#define TM 128
#define TN 64
#define TK 32

static __device__ __forceinline__ float sigf(float x) {
  return 1.0f / (1.0f + __expf(-x));
}

// ---------------------------------------------------------------------------
// Generalized strided / batched GEMM using v_wmma_f32_16x16x32_f16.
//   C[z](m,n) = act(alpha * sum_k A[z](m,k)*B[z](k,n) + bias[n % biasMod])
//   element(m,k) of A at A + zoff + m*rsA + k*csA  (zoff = (z/divA)*sA1 + (z%divA)*sA2)
//
// 256 threads = 8 wave32s; block tile 128x64; each wave owns 2 row-strips x
// 2 col-tiles -> 4 WMMAs per K-tile from 4 fragments (8 ds_load_b128).
//
// LDS tiles stored in *fragment order*: within each A row (B column) the 32
// K-halves are permuted by p(k) = ((k>>3)&1)*16 + ((k>>4)&1)*8 + (k&7), so each
// lane's 16 fragment halves are a contiguous 32B chunk -> ds_load_b128 pairs.
//
// OOB policy: A rows >= M and B cols >= N are *clamped, not zeroed* (they only
// feed D rows/cols that are never stored). Only the K-tail tile zero-fills.
// Pipeline is explicitly peeled: the main loop body has no conditionals; the
// last full tile and the K-tail are handled outside. LDS double-buffered,
// one barrier per tile.
// ---------------------------------------------------------------------------
__global__ __launch_bounds__(256) void gemm_wmma_kernel(
    const float* __restrict__ A, const float* __restrict__ B,
    const float* __restrict__ bias, float* __restrict__ C,
    int M, int N, int K,
    long rsA, long csA, long rsB, long csB, long rsC, long csC,
    int divA, long sA1, long sA2,
    int divB, long sB1, long sB2,
    int divC, long sC1, long sC2,
    float alpha, int biasMod, int actMode)
{
  __shared__ alignas(64) _Float16 Asp[2][TM * TK];   // 2 x 8KB
  __shared__ alignas(64) _Float16 Bsp[2][TN * TK];   // 2 x 4KB

  const int z = (int)blockIdx.z;
  A += (long)(z / divA) * sA1 + (long)(z % divA) * sA2;
  B += (long)(z / divB) * sB1 + (long)(z % divB) * sB2;
  C += (long)(z / divC) * sC1 + (long)(z % divC) * sC2;

  const int m0 = (int)blockIdx.y * TM;
  const int n0 = (int)blockIdx.x * TN;

  const int tid  = (int)threadIdx.x;
  const int wave = tid >> 5;
  const int lane = tid & 31;
  const int wr   = wave >> 1;   // 0..3  -> row strips wr and wr+4
  const int wc   = wave & 1;    // 0..1  -> 32-col strip (two 16-wide tiles)

  // staging coordinates: thread covers k-pair sp; A rows sr+16t (t=0..7),
  // B cols sr+16t (t=0..3)
  const int sr   = tid >> 4;               // 0..15
  const int sp   = (tid & 15) * 2;         // even k in 0..30
  const int pmap = ((sp >> 3) & 1) * 16 + ((sp >> 4) & 1) * 8 + (sp & 7);

  long aRowOff[8];                 // clamped row offsets (safe: see OOB policy)
  long bColOff[4];
  const float* pA[8];
  const float* pB[4];
#pragma unroll
  for (int t = 0; t < 8; ++t) {
    int r  = sr + 16 * t;
    int rg = m0 + r; if (rg > M - 1) rg = M - 1;
    aRowOff[t] = (long)rg * rsA;
    pA[t] = A + aRowOff[t] + (long)sp * csA;
  }
#pragma unroll
  for (int t = 0; t < 4; ++t) {
    int cg = n0 + sr + 16 * t; if (cg > N - 1) cg = N - 1;
    bColOff[t] = (long)cg * csB;
    pB[t] = B + (long)sp * rsB + bColOff[t];
  }
  const long stepA = (long)TK * csA;
  const long stepB = (long)TK * rsB;

  const int nFull = K / TK;
  const int rem   = K - nFull * TK;

  // ---- staging helpers ----
  auto stageFull = [&](int b) {
#pragma unroll
    for (int t = 0; t < 8; ++t) {
      float a0 = pA[t][0];
      float a1 = pA[t][csA];
      union { unsigned u; _Float16 h[2]; } pa;
      pa.h[0] = (_Float16)a0; pa.h[1] = (_Float16)a1;
      *(unsigned*)&Asp[b][(sr + 16 * t) * TK + pmap] = pa.u;
      pA[t] += stepA;
    }
#pragma unroll
    for (int t = 0; t < 4; ++t) {
      float b0 = pB[t][0];
      float b1 = pB[t][rsB];
      union { unsigned u; _Float16 h[2]; } pb;
      pb.h[0] = (_Float16)b0; pb.h[1] = (_Float16)b1;
      *(unsigned*)&Bsp[b][(sr + 16 * t) * TK + pmap] = pb.u;
      pB[t] += stepB;
    }
    __builtin_prefetch(pA[0], 0, 1);
  };
  auto stageTail = [&](int b) {
    const int ka  = nFull * TK + sp;
    const bool ok0 = (ka < K), ok1 = (ka + 1 < K);
    const long kA0 = (long)(ok0 ? ka : K - 1) * csA;
    const long kA1 = (long)(ok1 ? ka + 1 : K - 1) * csA;
    const long kB0 = (long)(ok0 ? ka : K - 1) * rsB;
    const long kB1 = (long)(ok1 ? ka + 1 : K - 1) * rsB;
#pragma unroll
    for (int t = 0; t < 8; ++t) {
      float a0 = A[aRowOff[t] + kA0];
      float a1 = A[aRowOff[t] + kA1];
      a0 = ok0 ? a0 : 0.0f;
      a1 = ok1 ? a1 : 0.0f;
      union { unsigned u; _Float16 h[2]; } pa;
      pa.h[0] = (_Float16)a0; pa.h[1] = (_Float16)a1;
      *(unsigned*)&Asp[b][(sr + 16 * t) * TK + pmap] = pa.u;
    }
#pragma unroll
    for (int t = 0; t < 4; ++t) {
      float b0 = B[kB0 + bColOff[t]];
      float b1 = B[kB1 + bColOff[t]];
      b0 = ok0 ? b0 : 0.0f;
      b1 = ok1 ? b1 : 0.0f;
      union { unsigned u; _Float16 h[2]; } pb;
      pb.h[0] = (_Float16)b0; pb.h[1] = (_Float16)b1;
      *(unsigned*)&Bsp[b][(sr + 16 * t) * TK + pmap] = pb.u;
    }
  };

  union { v8f v; float e[8]; } acc00, acc01, acc10, acc11;
#pragma unroll
  for (int i = 0; i < 8; ++i) {
    acc00.e[i] = 0.0f; acc01.e[i] = 0.0f; acc10.e[i] = 0.0f; acc11.e[i] = 0.0f;
  }

  const int g16  = (lane >> 4) * 16;
  const int rsel = lane & 15;
  const int aOff0 = (wr * 16 + rsel) * TK + g16;
  const int aOff1 = ((wr + 4) * 16 + rsel) * TK + g16;
  const int bOff0 = (wc * 32 + rsel) * TK + g16;
  const int bOff1 = (wc * 32 + 16 + rsel) * TK + g16;

  auto compute = [&](int b) {
    v16h af0 = *(const v16h*)&Asp[b][aOff0];
    v16h af1 = *(const v16h*)&Asp[b][aOff1];
    v16h bf0 = *(const v16h*)&Bsp[b][bOff0];
    v16h bf1 = *(const v16h*)&Bsp[b][bOff1];
    acc00.v = __builtin_amdgcn_wmma_f32_16x16x32_f16(
        false, af0, false, bf0, (short)0, acc00.v, false, false);
    acc01.v = __builtin_amdgcn_wmma_f32_16x16x32_f16(
        false, af0, false, bf1, (short)0, acc01.v, false, false);
    acc10.v = __builtin_amdgcn_wmma_f32_16x16x32_f16(
        false, af1, false, bf0, (short)0, acc10.v, false, false);
    acc11.v = __builtin_amdgcn_wmma_f32_16x16x32_f16(
        false, af1, false, bf1, (short)0, acc11.v, false, false);
  };

  // ---- explicitly peeled double-buffered pipeline ----
  if (nFull > 0) stageFull(0); else stageTail(0);
  __syncthreads();
  int buf = 0;
  for (int it = 0; it < nFull - 1; ++it) {       // branch-free steady state
    stageFull(buf ^ 1);
    compute(buf);
    __syncthreads();
    buf ^= 1;
  }
  if (nFull > 0) {                               // last full tile
    if (rem) stageTail(buf ^ 1);
    compute(buf);
    __syncthreads();
    buf ^= 1;
  }
  if (rem) {                                     // K-tail tile
    compute(buf);
  }

  // D layout: VGPR i -> row (strip*16 + (lane>=16)*8 + i), lane&15 -> col
  const int cc  = lane & 15;
  const int rhi = (lane >> 4) * 8;
#pragma unroll
  for (int i = 0; i < 8; ++i) {
    int rowA = m0 + wr * 16 + rhi + i;
    int rowB = rowA + 64;
    int col0 = n0 + wc * 32 + cc;
    int col1 = col0 + 16;
    float b0 = biasMod ? bias[col0 % biasMod] : 0.0f;
    float b1 = biasMod ? bias[col1 % biasMod] : 0.0f;
    if (rowA < M) {
      if (col0 < N) {
        float v = alpha * acc00.e[i] + b0;
        if (actMode == 1) v = fmaxf(v, 0.0f);
        C[(long)rowA * rsC + (long)col0 * csC] = v;
      }
      if (col1 < N) {
        float v = alpha * acc01.e[i] + b1;
        if (actMode == 1) v = fmaxf(v, 0.0f);
        C[(long)rowA * rsC + (long)col1 * csC] = v;
      }
    }
    if (rowB < M) {
      if (col0 < N) {
        float v = alpha * acc10.e[i] + b0;
        if (actMode == 1) v = fmaxf(v, 0.0f);
        C[(long)rowB * rsC + (long)col0 * csC] = v;
      }
      if (col1 < N) {
        float v = alpha * acc11.e[i] + b1;
        if (actMode == 1) v = fmaxf(v, 0.0f);
        C[(long)rowB * rsC + (long)col1 * csC] = v;
      }
    }
  }
}

// ---------------------------------------------------------------------------
// LayerNorm over last dim C; one block (128 thr) per row.
// ---------------------------------------------------------------------------
__global__ __launch_bounds__(128) void ln_kernel(
    const float* __restrict__ x, const float* __restrict__ g,
    const float* __restrict__ b, float* __restrict__ y, int C)
{
  long row = blockIdx.x;
  const float* xr = x + row * (long)C;
  float* yr = y + row * (long)C;
  float s = 0.f, s2 = 0.f;
  for (int c = threadIdx.x; c < C; c += 128) {
    float v = xr[c]; s += v; s2 += v * v;
  }
  __shared__ float rs[128], rs2[128];
  rs[threadIdx.x] = s; rs2[threadIdx.x] = s2; __syncthreads();
  for (int o = 64; o > 0; o >>= 1) {
    if ((int)threadIdx.x < o) { rs[threadIdx.x] += rs[threadIdx.x + o]; rs2[threadIdx.x] += rs2[threadIdx.x + o]; }
    __syncthreads();
  }
  float mean = rs[0] / C;
  float var  = rs2[0] / C - mean * mean;
  float inv  = rsqrtf(var + 1e-5f);
  for (int c = threadIdx.x; c < C; c += 128)
    yr[c] = (xr[c] - mean) * inv * g[c] + b[c];
}

// ---------------------------------------------------------------------------
// Softmax over k (256 wide rows) with pair bias (stored (q,k,h)) + mask bias.
// row index r: h = (r>>8)%H, q = r&255.
// ---------------------------------------------------------------------------
__global__ __launch_bounds__(256) void softmax_bias_kernel(
    float* __restrict__ sc, const float* __restrict__ bias,
    const float* __restrict__ mask, int H)
{
  long r = blockIdx.x;
  int k = (int)threadIdx.x;
  int q = (int)(r & 255);
  int h = (int)((r >> 8) % H);
  float v = sc[r * 256 + k] + bias[(((long)q * 256 + k) * H) + h]
          + (mask[k] - 1.0f) * 1e9f;
  __shared__ float red[256];
  red[k] = v; __syncthreads();
  for (int o = 128; o > 0; o >>= 1) { if (k < o) red[k] = fmaxf(red[k], red[k + o]); __syncthreads(); }
  float mx = red[0]; __syncthreads();
  float e = __expf(v - mx);
  red[k] = e; __syncthreads();
  for (int o = 128; o > 0; o >>= 1) { if (k < o) red[k] += red[k + o]; __syncthreads(); }
  sc[r * 256 + k] = e / red[0];
}

// ---------------------------------------------------------------------------
// Elementwise kernels (grid-stride).
// ---------------------------------------------------------------------------
__global__ void add_kernel(float* __restrict__ y, const float* __restrict__ x, long n) {
  for (long i = blockIdx.x * 256L + threadIdx.x; i < n; i += gridDim.x * 256L) y[i] += x[i];
}
__global__ void sigmul_kernel(float* __restrict__ y, const float* __restrict__ g, long n) {
  for (long i = blockIdx.x * 256L + threadIdx.x; i < n; i += gridDim.x * 256L) y[i] *= sigf(g[i]);
}
__global__ void add_sigmul_kernel(float* __restrict__ y, const float* __restrict__ x,
                                  const float* __restrict__ g, long n) {
  for (long i = blockIdx.x * 256L + threadIdx.x; i < n; i += gridDim.x * 256L) y[i] += x[i] * sigf(g[i]);
}
__global__ void trimul_gate_kernel(float* __restrict__ p, const float* __restrict__ g,
                                   const float* __restrict__ mask, long n) {
  for (long idx = blockIdx.x * 256L + threadIdx.x; idx < n; idx += gridDim.x * 256L) {
    long ij = idx >> 7;               // / PAIRC
    int j = (int)(ij & 255);
    int i = (int)(ij >> 8);
    p[idx] *= sigf(g[idx]) * mask[i] * mask[j];
  }
}
__global__ void rowscale_kernel(float* __restrict__ x, const float* __restrict__ mask, int C, long n) {
  for (long i = blockIdx.x * 256L + threadIdx.x; i < n; i += gridDim.x * 256L) x[i] *= mask[i / C];
}
__global__ void transpose_pair_kernel(float* __restrict__ dst, const float* __restrict__ src) {
  const long n = 256L * 256 * PAIRC;
  for (long idx = blockIdx.x * 256L + threadIdx.x; idx < n; idx += gridDim.x * 256L) {
    int c = (int)(idx & (PAIRC - 1));
    long ij = idx >> 7;
    int j = (int)(ij & 255);
    int i = (int)(ij >> 8);
    dst[(((long)j * 256 + i) << 7) + c] = src[idx];
  }
}
__global__ void add_transpose_kernel(float* __restrict__ pair, const float* __restrict__ d) {
  const long n = 256L * 256 * PAIRC;
  for (long idx = blockIdx.x * 256L + threadIdx.x; idx < n; idx += gridDim.x * 256L) {
    int c = (int)(idx & (PAIRC - 1));
    long ij = idx >> 7;
    int j = (int)(ij & 255);
    int i = (int)(ij >> 8);
    pair[idx] += d[(((long)j * 256 + i) << 7) + c];
  }
}
__global__ void seq_embed_kernel(const int* __restrict__ seq, const float* __restrict__ aa,
                                 float* __restrict__ out) {
  int l = blockIdx.x;
  int t = seq[l];
  for (int c = threadIdx.x; c < SEQC; c += 128)
    out[(long)l * SEQC + c] = aa[(long)t * SEQC + c];
}
__global__ void pair_embed_kernel(const int* __restrict__ residx, const float* __restrict__ relpos,
                                  float* __restrict__ out) {
  long ij = blockIdx.x;
  int i = (int)(ij >> 8), j = (int)(ij & 255);
  int pi = residx[i + 1] - 1;
  int pj = residx[j + 1] - 1;
  int off = pj - pi;
  int rel = off + MAXREL;
  rel = rel < 0 ? 0 : (rel > 2 * MAXREL ? 2 * MAXREL : rel);
  rel += 1;
  for (int c = threadIdx.x; c < PAIRC; c += 128)
    out[ij * PAIRC + c] = relpos[(long)rel * PAIRC + c];
}

// ---------------------------------------------------------------------------
// Host-side orchestration
// ---------------------------------------------------------------------------
struct LNp  { const float *g, *b; };
struct Lin  { const float *w, *b; };
struct TriMult { LNp ln_in, ln_out; Lin left, right, left_gate, right_gate, out, gate; };
struct TriAttn { LNp ln; const float* bias_w; const float *wq, *wk, *wv; Lin gate, out; };
struct Block {
  LNp sa_ln, sa_ln2d; const float* sa_bias_w; const float *sa_wq, *sa_wk, *sa_wv; Lin sa_gate, sa_out;
  LNp st_ln; Lin st_l1, st_l2;
  LNp opm_ln; Lin opm_left, opm_right, opm_out;
  TriMult tm_out, tm_in;
  TriAttn ta_start, ta_end;
  LNp pt_ln; Lin pt_l1, pt_l2;
};

extern "C" void kernel_launch(void* const* d_in, const int* in_sizes, int n_in,
                              void* d_out, int out_size, void* d_ws, size_t ws_size,
                              hipStream_t stream) {
  (void)in_sizes; (void)n_in; (void)out_size; (void)ws_size;

  const int*   seq_ids = (const int*)d_in[0];
  const int*   residx  = (const int*)d_in[1];
  const float* mask    = (const float*)d_in[2];
  int cur = 3;
  auto F = [&]() { return (const float*)d_in[cur++]; };
  const float* aa_embed = F();
  const float* relpos   = F();
  auto getLN  = [&](LNp& l) { l.g = F(); l.b = F(); };
  auto getLin = [&](Lin& l) { l.w = F(); l.b = F(); };
  auto getTM  = [&](TriMult& t) { getLN(t.ln_in); getLN(t.ln_out); getLin(t.left); getLin(t.right);
                                  getLin(t.left_gate); getLin(t.right_gate); getLin(t.out); getLin(t.gate); };
  auto getTA  = [&](TriAttn& t) { getLN(t.ln); t.bias_w = F(); t.wq = F(); t.wk = F(); t.wv = F();
                                  getLin(t.gate); getLin(t.out); };

  Block blocks[2];
  for (int bi = 0; bi < 2; ++bi) {
    Block& bk = blocks[bi];
    getLN(bk.sa_ln); getLN(bk.sa_ln2d); bk.sa_bias_w = F();
    bk.sa_wq = F(); bk.sa_wk = F(); bk.sa_wv = F(); getLin(bk.sa_gate); getLin(bk.sa_out);
    getLN(bk.st_ln); getLin(bk.st_l1); getLin(bk.st_l2);
    getLN(bk.opm_ln); getLin(bk.opm_left); getLin(bk.opm_right); getLin(bk.opm_out);
    getTM(bk.tm_out); getTM(bk.tm_in);
    getTA(bk.ta_start); getTA(bk.ta_end);
    getLN(bk.pt_ln); getLin(bk.pt_l1); getLin(bk.pt_l2);
  }

  // live activations in d_out (tuple: seq_act then pair_act, flat f32)
  float* seq_act  = (float*)d_out;
  float* pair_act = seq_act + (long)LSEQ * SEQC;

  // workspace layout
  const long P = 256L * 256 * PAIRC;            // pair-sized buffer (8.39M floats)
  float* ws = (float*)d_ws;
  float* t0 = ws + 0 * P;
  float* t1 = ws + 1 * P;
  float* t2 = ws + 2 * P;
  float* t3 = ws + 3 * P;
  float* t4 = ws + 4 * P;
  float* t5 = ws + 5 * P;
  float* SC = ws + 6 * P;                       // score chunk / transition hidden buffer
  float* sa0 = ws + 7 * P;                      // seq scratch: 4 x 256*1536
  float* sa1 = sa0 + 256L * 1536;
  float* sa2 = sa1 + 256L * 1536;
  float* sa3 = sa2 + 256L * 1536;

  auto blocks1d = [](long n) { long b = (n + 255) / 256; if (b > 4096) b = 4096; return dim3((unsigned)b); };

  auto gemm = [&](const float* A, const float* B, const float* bias, float* C,
                  int M, int N, int K,
                  long rsA, long csA, long rsB, long csB, long rsC, long csC,
                  int batches,
                  int divA, long sA1, long sA2,
                  int divB, long sB1, long sB2,
                  int divC, long sC1, long sC2,
                  float alpha, int biasMod, int act) {
    dim3 g((N + TN - 1) / TN, (M + TM - 1) / TM, batches);
    gemm_wmma_kernel<<<g, 256, 0, stream>>>(A, B, bias, C, M, N, K,
        rsA, csA, rsB, csB, rsC, csC,
        divA, sA1, sA2, divB, sB1, sB2, divC, sC1, sC2, alpha, biasMod, act);
  };
  // plain row-major linear: C(MxN) = A(MxK) @ W(KxN) + b
  auto lin = [&](const float* A, const float* W, const float* b, float* C,
                 int M, int N, int K, int act = 0) {
    gemm(A, W, b, C, M, N, K, K, 1, N, 1, N, 1, 1, 1, 0, 0, 1, 0, 0, 1, 0, 0, 1.0f, b ? N : 0, act);
  };
  auto ln = [&](const float* x, const LNp& p, float* y, long rows, int C) {
    ln_kernel<<<dim3((unsigned)rows), 128, 0, stream>>>(x, p.g, p.b, y, C);
  };
  auto add = [&](float* y, const float* x, long n) {
    add_kernel<<<blocks1d(n), 256, 0, stream>>>(y, x, n);
  };

  // ---------------- embeddings ----------------
  seq_embed_kernel<<<LSEQ, 128, 0, stream>>>(seq_ids, aa_embed, seq_act);
  pair_embed_kernel<<<LSEQ * LSEQ, 128, 0, stream>>>(residx, relpos, pair_act);

  const float seq_scale = 0.14433756729740643f;   // 48^-0.5
  const float tri_scale = 0.17677669529663687f;   // 32^-0.5

  for (int bi = 0; bi < 2; ++bi) {
    const Block& bk = blocks[bi];

    // ---------------- seq attention ----------------
    ln(seq_act, bk.sa_ln, sa0, LSEQ, SEQC);
    ln(pair_act, bk.sa_ln2d, t0, 65536, PAIRC);
    // pair bias (q,k,h): (L*L x 8) = pa @ bias_w
    gemm(t0, bk.sa_bias_w, nullptr, t1, 65536, HSEQ, PAIRC,
         PAIRC, 1, HSEQ, 1, HSEQ, 1, 1, 1, 0, 0, 1, 0, 0, 1, 0, 0, 1.0f, 0, 0);
    lin(sa0, bk.sa_wq, nullptr, sa1, LSEQ, SEQC, SEQC);
    lin(sa0, bk.sa_wk, nullptr, sa2, LSEQ, SEQC, SEQC);
    lin(sa0, bk.sa_wv, nullptr, sa3, LSEQ, SEQC, SEQC);
    // scores per head: (h,q,k) in t0
    gemm(sa1, sa2, nullptr, t0, LSEQ, LSEQ, DSEQ,
         SEQC, 1, 1, SEQC, LSEQ, 1, HSEQ,
         1, DSEQ, 0, 1, DSEQ, 0, 1, 65536, 0, seq_scale, 0, 0);
    softmax_bias_kernel<<<HSEQ * LSEQ, 256, 0, stream>>>(t0, t1, mask, HSEQ);
    // o = scores @ v -> (q,h,d) packed as (q,c) in sa1
    gemm(t0, sa3, nullptr, sa1, LSEQ, DSEQ, LSEQ,
         LSEQ, 1, SEQC, 1, SEQC, 1, HSEQ,
         1, 65536, 0, 1, DSEQ, 0, 1, DSEQ, 0, 1.0f, 0, 0);
    lin(sa0, bk.sa_gate.w, bk.sa_gate.b, sa2, LSEQ, SEQC, SEQC);
    sigmul_kernel<<<blocks1d(LSEQ * (long)SEQC), 256, 0, stream>>>(sa1, sa2, LSEQ * (long)SEQC);
    lin(sa1, bk.sa_out.w, bk.sa_out.b, sa2, LSEQ, SEQC, SEQC);
    add(seq_act, sa2, LSEQ * (long)SEQC);

    // ---------------- seq transition (ReLU fused in GEMM epilogue) ----------------
    ln(seq_act, bk.st_ln, sa0, LSEQ, SEQC);
    lin(sa0, bk.st_l1.w, bk.st_l1.b, sa1, LSEQ, 4 * SEQC, SEQC, 1);
    lin(sa1, bk.st_l2.w, bk.st_l2.b, sa2, LSEQ, SEQC, 4 * SEQC);
    add(seq_act, sa2, LSEQ * (long)SEQC);

    // ---------------- outer product mean ----------------
    ln(seq_act, bk.opm_ln, sa0, LSEQ, SEQC);
    lin(sa0, bk.opm_left.w, bk.opm_left.b, sa1, LSEQ, OPMH, SEQC);
    rowscale_kernel<<<blocks1d(LSEQ * (long)OPMH), 256, 0, stream>>>(sa1, mask, OPMH, LSEQ * (long)OPMH);
    lin(sa0, bk.opm_right.w, bk.opm_right.b, sa2, LSEQ, OPMH, SEQC);
    rowscale_kernel<<<blocks1d(LSEQ * (long)OPMH), 256, 0, stream>>>(sa2, mask, OPMH, LSEQ * (long)OPMH);
    // T[c](j,p) = sum_d right(j,d) * Wout(c*32+d, p)    batch c=0..31 -> t4
    gemm(sa2, bk.opm_out.w, nullptr, t4, LSEQ, PAIRC, OPMH,
         OPMH, 1, PAIRC, 1, PAIRC, 1, OPMH,
         1, 0, 0, 1, (long)OPMH * PAIRC, 0, 1, (long)LSEQ * PAIRC, 0, 1.0f, 0, 0);
    // O(i, j*128+p) = sum_c left(i,c) * T(c, j*128+p)  -> t0; bias over p
    gemm(sa1, t4, bk.opm_out.b, t0, LSEQ, LSEQ * PAIRC, OPMH,
         OPMH, 1, (long)LSEQ * PAIRC, 1, (long)LSEQ * PAIRC, 1, 1,
         1, 0, 0, 1, 0, 0, 1, 0, 0, 1.0f, PAIRC, 0);
    add(pair_act, t0, P);

    // ---------------- triangle multiplications ----------------
    for (int tm = 0; tm < 2; ++tm) {
      const TriMult& p = (tm == 0) ? bk.tm_out : bk.tm_in;
      bool outgoing = (tm == 0);
      ln(pair_act, p.ln_in, t0, 65536, PAIRC);
      lin(t0, p.left.w, p.left.b, t1, 65536, PAIRC, PAIRC);
      lin(t0, p.left_gate.w, p.left_gate.b, t2, 65536, PAIRC, PAIRC);
      trimul_gate_kernel<<<blocks1d(P), 256, 0, stream>>>(t1, t2, mask, P);
      lin(t0, p.right.w, p.right.b, t2, 65536, PAIRC, PAIRC);
      lin(t0, p.right_gate.w, p.right_gate.b, t3, 65536, PAIRC, PAIRC);
      trimul_gate_kernel<<<blocks1d(P), 256, 0, stream>>>(t2, t3, mask, P);
      // einsum, batched over channel c (128 GEMMs of 256x256x256) -> t3
      if (outgoing) {
        // O_c(i,j) = sum_k L(i,k,c) R(j,k,c)
        gemm(t1, t2, nullptr, t3, LSEQ, LSEQ, LSEQ,
             (long)LSEQ * PAIRC, PAIRC, PAIRC, (long)LSEQ * PAIRC,
             (long)LSEQ * PAIRC, PAIRC, PAIRC,
             1, 1, 0, 1, 1, 0, 1, 1, 0, 1.0f, 0, 0);
      } else {
        // O_c(i,j) = sum_k L(k,i,c) R(k,j,c)
        gemm(t1, t2, nullptr, t3, LSEQ, LSEQ, LSEQ,
             PAIRC, (long)LSEQ * PAIRC, (long)LSEQ * PAIRC, PAIRC,
             (long)LSEQ * PAIRC, PAIRC, PAIRC,
             1, 1, 0, 1, 1, 0, 1, 1, 0, 1.0f, 0, 0);
      }
      ln(t3, p.ln_out, t4, 65536, PAIRC);
      lin(t4, p.out.w, p.out.b, t5, 65536, PAIRC, PAIRC);
      lin(t0, p.gate.w, p.gate.b, t4, 65536, PAIRC, PAIRC);
      add_sigmul_kernel<<<blocks1d(P), 256, 0, stream>>>(pair_act, t5, t4, P);
    }

    // ---------------- triangle attentions ----------------
    for (int ta = 0; ta < 2; ++ta) {
      const TriAttn& p = (ta == 0) ? bk.ta_start : bk.ta_end;
      bool starting = (ta == 0);
      const float* src = pair_act;
      if (!starting) {
        transpose_pair_kernel<<<blocks1d(P), 256, 0, stream>>>(t5, pair_act);
        src = t5;
      }
      ln(src, p.ln, t0, 65536, PAIRC);
      // bias (q,k,h) -> t1
      gemm(t0, p.bias_w, nullptr, t1, 65536, HPAIR, PAIRC,
           PAIRC, 1, HPAIR, 1, HPAIR, 1, 1, 1, 0, 0, 1, 0, 0, 1, 0, 0, 1.0f, 0, 0);
      lin(t0, p.wq, nullptr, t2, 65536, PAIRC, PAIRC);
      lin(t0, p.wk, nullptr, t3, 65536, PAIRC, PAIRC);
      lin(t0, p.wv, nullptr, t4, 65536, PAIRC, PAIRC);
      // chunk over i (32 rows): batch z = i'*4 + h  (128 per chunk)
      const long ROWS = (long)LSEQ * PAIRC;       // 32768 floats per i-row
      for (int ci = 0; ci < 8; ++ci) {
        const float* Aq = t2 + (long)ci * 32 * ROWS;
        const float* Bk = t3 + (long)ci * 32 * ROWS;
        const float* Bv = t4 + (long)ci * 32 * ROWS;
        float* Oo = t5 + (long)ci * 32 * ROWS;
        // scores(z,q,k) -> SC
        gemm(Aq, Bk, nullptr, SC, LSEQ, LSEQ, DPAIR,
             PAIRC, 1, 1, PAIRC, LSEQ, 1, 32 * HPAIR,
             HPAIR, ROWS, DPAIR, HPAIR, ROWS, DPAIR, 1, 65536, 0, tri_scale, 0, 0);
        softmax_bias_kernel<<<32 * HPAIR * LSEQ, 256, 0, stream>>>(SC, t1, mask, HPAIR);
        // o = scores @ v -> (i,q,h,d) packed (i,q,c)
        gemm(SC, Bv, nullptr, Oo, LSEQ, DPAIR, LSEQ,
             LSEQ, 1, PAIRC, 1, PAIRC, 1, 32 * HPAIR,
             1, 65536, 0, HPAIR, ROWS, DPAIR, HPAIR, ROWS, DPAIR, 1.0f, 0, 0);
      }
      lin(t0, p.gate.w, p.gate.b, t2, 65536, PAIRC, PAIRC);
      sigmul_kernel<<<blocks1d(P), 256, 0, stream>>>(t5, t2, P);
      lin(t5, p.out.w, p.out.b, t3, 65536, PAIRC, PAIRC);
      if (starting) add(pair_act, t3, P);
      else add_transpose_kernel<<<blocks1d(P), 256, 0, stream>>>(pair_act, t3);
    }

    // ---------------- pair transition (chunked; ReLU fused) ----------------
    ln(pair_act, bk.pt_ln, t0, 65536, PAIRC);
    for (int ch = 0; ch < 4; ++ch) {
      float* aCh = t0 + (long)ch * 16384 * PAIRC;
      float* oCh = t1 + (long)ch * 16384 * PAIRC;
      lin(aCh, bk.pt_l1.w, bk.pt_l1.b, SC, 16384, 4 * PAIRC, PAIRC, 1);
      lin(SC, bk.pt_l2.w, bk.pt_l2.b, oCh, 16384, PAIRC, 4 * PAIRC);
    }
    add(pair_act, t1, P);
  }
}
